// PConv_BN_Merge_4690104287496
// MI455X (gfx1250) — matrix operand order repaired and checked by reading, not compiled, not run on hardware
//
#include <hip/hip_runtime.h>
#include <hip/hip_bf16.h>

typedef __attribute__((ext_vector_type(16))) _Float16 v16h;
typedef __attribute__((ext_vector_type(8)))  float    v8f;

#define COUT    256
#define GRPS    4
#define CIN_G   64
#define HH      32
#define WW_     32
#define BB      32
#define NSP     32768          // B*H*W
#define KTOT    576            // 64*9
#define KTILES  18             // 576/32
#define MTILES  64             // 1024/16

// ---- workspace layout (bytes) ----
// psum  f32 [1024][32768]               : 134,217,728
// Apack f16 [64 mt][18 kt][32 lane][16] :   1,179,648
// inv   f32 [1024], shift f32 [1024]    :       8,192
static const size_t PSUM_OFF  = 0;
static const size_t APACK_OFF = (size_t)1024 * 32768 * 4;
static const size_t BN_OFF    = APACK_OFF + (size_t)MTILES * KTILES * 32 * 16 * 2;

// ---------------------------------------------------------------------------
// Kernel 1: alpha = mean|w| per output channel; pack sign(w)*alpha as f16 in
// the exact WMMA A-fragment (16x32 f16) lane layout so the conv kernel loads
// one contiguous v16h per lane per K-step.
//   lane<16 : M=lane,    h 0..7 -> K=h,     h 8..15 -> K=h+8   (K 0-7,16-23)
//   lane>=16: M=lane-16, h 0..7 -> K=h+8,   h 8..15 -> K=h+16  (K 8-15,24-31)
// ---------------------------------------------------------------------------
__global__ __launch_bounds__(256) void prep_weights(const float* __restrict__ w,
                                                    _Float16* __restrict__ Apack) {
  const int co = blockIdx.x;
  const int t  = threadIdx.x;
  __shared__ float red[256];
  const float* wc = w + (size_t)co * 2304;   // [256][3][3] contiguous
  float s = 0.f;
  for (int e = t; e < 2304; e += 256) s += fabsf(wc[e]);
  red[t] = s; __syncthreads();
  for (int st = 128; st > 0; st >>= 1) { if (t < st) red[t] += red[t + st]; __syncthreads(); }
  const float alpha = red[0] * (1.0f / 2304.0f);
  __syncthreads();

  const int m_local = co & 15;
  const int mt_loc  = co >> 4;               // tile within group
  for (int e = t; e < 2304; e += 256) {
    const float wv = wc[e];
    const int cin = e / 9, kp = e - cin * 9; // e = cin*9 + (kh*3+kw)
    const int g = cin >> 6, ci = cin & 63;
    const int k  = ci * 9 + kp;              // K index inside group: ci*9 + kpos
    const int kt = k >> 5, kk = k & 31;
    int lane, h;
    if      (kk <  8) { lane = m_local;      h = kk;      }
    else if (kk < 16) { lane = m_local + 16; h = kk - 8;  }
    else if (kk < 24) { lane = m_local;      h = kk - 8;  }
    else              { lane = m_local + 16; h = kk - 16; }
    const int mt = g * 16 + mt_loc;          // 0..63 over all 1024 psum channels
    const size_t idx = (((size_t)(mt * KTILES + kt)) * 32 + lane) * 16 + h;
    Apack[idx] = (_Float16)((wv >= 0.f) ? alpha : -alpha);
  }
}

// ---------------------------------------------------------------------------
// Kernel 2: implicit-GEMM grouped conv with v_wmma_f32_16x16x32_f16.
// Block = 256 thr (8 waves). grid.x = 8 channel-blocks (128 ch each, whole
// block inside one group), grid.y = 1024 spatial tiles of 32 (= one full
// image row, so w0 == 0 and the conv window is ww = n + kw - 1).
// Each wave owns one 16-channel tile and TWO independent 16x16 C tiles
// (spatial columns 0-15 and 16-31) sharing each A fragment -> two
// independent WMMA chains for pipelining, half the A traffic per output.
// LDS B layout [kt][n(32)][kk(32)] f16: lane reads 16 contiguous halves at
// column n, half-range selected by lane/16 (B 32x16 f16 fragment layout).
// ---------------------------------------------------------------------------
__global__ __launch_bounds__(256) void conv_wmma(const float* __restrict__ x,
                                                 const _Float16* __restrict__ Apack,
                                                 float* __restrict__ psum) {
  __shared__ _Float16 Blds[KTILES * 1024];   // 36 KB: [kt][n 0..31][kk 0..31]
  const int t    = threadIdx.x;
  const int wave = t >> 5, lane = t & 31;
  const int cb = blockIdx.x;                 // 0..7
  const int g  = cb >> 1;                    // CiM group
  const int mt = cb * 8 + wave;              // 16-channel tile, 0..63
  const int s0 = blockIdx.y * 32;            // spatial base (row aligned)
  const int b  = s0 >> 10;
  const int h  = (s0 >> 5) & 31;
  const float* xg = x + ((size_t)b * 256 + g * 64) * 1024;

  // stage im2col tile: 576 K x 32 N, zero-padded SAME borders
  for (int e = t; e < KTOT * 32; e += 256) {
    const int k = e >> 5, n = e & 31;
    const int ci = k / 9, kp = k - ci * 9;
    const int kh = kp / 3, kw = kp - kh * 3;
    const int hh = h + kh - 1;
    const int ww = n + kw - 1;               // w0 == 0
    float v = 0.f;
    if (hh >= 0 && hh < HH && ww >= 0 && ww < WW_)
      v = xg[(size_t)ci * 1024 + hh * 32 + ww];
    const int kt = k >> 5, kk = k & 31;
    Blds[kt * 1024 + n * 32 + ((kk >> 3) & 1) * 16 + (kk >> 4) * 8 + (kk & 7)] = (_Float16)v;
  }
  __syncthreads();

  v8f c0 = {}, c1 = {};
  const v16h* Ap = (const v16h*)Apack + (size_t)mt * KTILES * 32 + lane;
  const _Float16* Bp0 = Blds + (lane & 15) * 32 + (lane >> 4) * 16;  // cols 0-15
  const _Float16* Bp1 = Bp0 + 16 * 32;                               // cols 16-31
  #pragma unroll
  for (int kt = 0; kt < KTILES; ++kt) {
    v16h a  = Ap[(size_t)kt * 32];
    v16h b0 = *(const v16h*)(Bp0 + kt * 1024);
    v16h b1 = *(const v16h*)(Bp1 + kt * 1024);
    c0 = __builtin_amdgcn_wmma_f32_16x16x32_f16(false, a, false, b0,
                                                (short)0, c0, false, false);
    c1 = __builtin_amdgcn_wmma_f32_16x16x32_f16(false, a, false, b1,
                                                (short)0, c1, false, false);
  }

  // C layout: VGPR r, lane -> M = r + 8*(lane>=16), N = lane%16
  const int ncol = lane & 15;
  const int mrow = (lane >> 4) * 8;
  float* pbase = psum + (size_t)(mt * 16 + mrow) * NSP + (s0 + ncol);
  #pragma unroll
  for (int r = 0; r < 8; ++r) {
    pbase[(size_t)r * NSP]      = c0[r];
    pbase[(size_t)r * NSP + 16] = c1[r];
  }
}

// ---------------------------------------------------------------------------
// Kernel 3: per-channel batch stats over 32768 values -> BN affine params.
// ---------------------------------------------------------------------------
__global__ __launch_bounds__(256) void bn_stats(const float* __restrict__ psum,
                                                const float* __restrict__ gamma,
                                                const float* __restrict__ beta,
                                                float* __restrict__ inv,
                                                float* __restrict__ shift) {
  const int ch = blockIdx.x;
  const int t  = threadIdx.x;
  __shared__ float rs[256], rq[256];
  const float* p = psum + (size_t)ch * NSP;
  float s = 0.f, q = 0.f;
  for (int i = t; i < NSP; i += 256) { const float v = p[i]; s += v; q += v * v; }
  rs[t] = s; rq[t] = q; __syncthreads();
  for (int st = 128; st > 0; st >>= 1) {
    if (t < st) { rs[t] += rs[t + st]; rq[t] += rq[t + st]; }
    __syncthreads();
  }
  if (t == 0) {
    const float mean = rs[0] * (1.0f / NSP);
    const float var  = rq[0] * (1.0f / NSP) - mean * mean;
    const float iv   = rsqrtf(var + 1e-5f) * gamma[ch];
    inv[ch]   = iv;
    shift[ch] = beta[ch] - mean * iv;
  }
}

// ---------------------------------------------------------------------------
// Kernel 4: BN affine + hardtanh/sign (sign survives the clip; sign(0)=+1)
// + merge 4 groups + 4-bit qrelu (rintf == round-half-even, matches jnp.round)
// ---------------------------------------------------------------------------
__global__ __launch_bounds__(256) void finalize(const float* __restrict__ psum,
                                                const float* __restrict__ inv,
                                                const float* __restrict__ shift,
                                                float* __restrict__ out) {
  const int idx = blockIdx.x * 256 + threadIdx.x;  // over 32*256*32*32
  const int hw  = idx & 1023;
  const int tmp = idx >> 10;
  const int co  = tmp & 255;
  const int b   = tmp >> 8;
  const int s   = b * 1024 + hw;
  float acc = 0.f;
  #pragma unroll
  for (int g = 0; g < GRPS; ++g) {
    const int ch  = g * 256 + co;
    const float v = psum[(size_t)ch * NSP + s] * inv[ch] + shift[ch];
    acc += (v >= 0.f) ? 1.f : -1.f;
  }
  const float u = fminf(fmaxf(acc * 0.25f, 0.f), 1.f);
  out[idx] = rintf(u * 15.f) * (4.f / 15.f);
}

extern "C" void kernel_launch(void* const* d_in, const int* in_sizes, int n_in,
                              void* d_out, int out_size, void* d_ws, size_t ws_size,
                              hipStream_t stream) {
  const float* x     = (const float*)d_in[0];
  const float* w     = (const float*)d_in[1];
  const float* gamma = (const float*)d_in[2];
  const float* beta  = (const float*)d_in[3];
  float* out = (float*)d_out;

  char* ws = (char*)d_ws;
  float*    psum  = (float*)(ws + PSUM_OFF);
  _Float16* Apack = (_Float16*)(ws + APACK_OFF);
  float*    inv   = (float*)(ws + BN_OFF);
  float*    shift = inv + 1024;

  prep_weights<<<COUT, 256, 0, stream>>>(w, Apack);

  dim3 grid2(8, NSP / 32);
  conv_wmma<<<grid2, 256, 0, stream>>>(x, Apack, psum);

  bn_stats<<<1024, 256, 0, stream>>>(psum, gamma, beta, inv, shift);

  finalize<<<(BB * COUT * HH * WW_) / 256, 256, 0, stream>>>(psum, inv, shift, out);
}
